// CausalSelfAttention_29978871726081
// MI455X (gfx1250) — compile-verified, matrix operand order checked by reading, not run on hardware
//
#include <hip/hip_runtime.h>
#include <hip/hip_bf16.h>
#include <stdint.h>

// CDNA5 / gfx1250: wave32, WMMA f32_16x16x32_bf16 everywhere,
// TDM (tensor_load_to_lds + TENSORcnt) double-buffered staging in the GEMMs.
typedef __bf16 bf16_t;
typedef __attribute__((ext_vector_type(16))) __bf16 v16bf;
typedef __attribute__((ext_vector_type(8)))  float  v8f;
typedef __attribute__((ext_vector_type(4)))  unsigned int v4u;
typedef __attribute__((ext_vector_type(8)))  int   v8i;
typedef __attribute__((ext_vector_type(4)))  int   v4i;

#define BB 4
#define TT 2048
#define CC 1024
#define HH 16
#define DD 64
#define MM (BB * TT)   // 8192 rows for the projection GEMMs

__device__ __forceinline__ bf16_t f2bf(float f) {
  unsigned u = __builtin_bit_cast(unsigned, f);
  unsigned r = (u + 0x7FFFu + ((u >> 16) & 1u)) >> 16;   // round-to-nearest-even
  return __builtin_bit_cast(bf16_t, (unsigned short)r);
}

// ---------------------------------------------------------------------------
// Tensor Data Mover: DMA a rows x cols bf16 tile (row stride in elements)
// from global memory into LDS.  D# built per CDNA5 ISA section 8.
// ---------------------------------------------------------------------------
__device__ __forceinline__ void tdm_load_tile_bf16(const bf16_t* gsrc,
                                                   unsigned int lds_off,
                                                   int rows, int cols,
                                                   int row_stride) {
  unsigned long long ga = (unsigned long long)(uintptr_t)gsrc;
  v4u g0 = {};
  g0[0] = 1u;                                             // count=1, user mode
  g0[1] = lds_off;                                        // LDS byte address
  g0[2] = (unsigned int)ga;                               // global_addr[31:0]
  g0[3] = (unsigned int)((ga >> 32) & 0x01FFFFFFu) | (2u << 30);  // [56:32], type=2

  unsigned int td0 = (unsigned int)cols;                  // tensor_dim0
  unsigned int td1 = (unsigned int)rows;                  // tensor_dim1
  unsigned int st0 = (unsigned int)row_stride;            // tensor_dim0_stride
  v8i g1 = {};
  g1[0] = (int)(1u << 16);                                // data_size=1 -> 2 bytes
  g1[1] = (int)((td0 & 0xFFFFu) << 16);                   // tensor_dim0[15:0]
  g1[2] = (int)((td0 >> 16) | ((td1 & 0xFFFFu) << 16));   // td0[31:16] | td1[15:0]
  g1[3] = (int)((td1 >> 16) | (((unsigned)cols) << 16));  // td1[31:16] | tile_dim0
  g1[4] = (int)((unsigned)rows & 0xFFFFu);                // tile_dim1 (tile_dim2=0)
  g1[5] = (int)st0;                                       // dim0 stride [31:0]
  g1[6] = 0;
  g1[7] = 0;

  v4i gz = {};
#if __clang_major__ >= 23
  v8i gz8 = {};
  __builtin_amdgcn_tensor_load_to_lds(g0, g1, gz, gz, gz8, 0);
#else
  __builtin_amdgcn_tensor_load_to_lds(g0, g1, gz, gz, 0);
#endif
}

// ---------------------------------------------------------------------------
// One-time conversions (O(data), negligible vs compute)
// ---------------------------------------------------------------------------
__global__ void f32_to_bf16_kernel(const float* __restrict__ in,
                                   bf16_t* __restrict__ out, int n) {
  int i = blockIdx.x * blockDim.x + threadIdx.x;
  int stride = gridDim.x * blockDim.x;
  for (; i < n; i += stride) out[i] = f2bf(in[i]);
}

// W (KxN row-major, f32) -> Wt (NxK row-major, bf16) so GEMM B tiles are
// plain 2D row-major tiles the TDM can stage directly.
__global__ void f32_to_bf16_T_kernel(const float* __restrict__ in,
                                     bf16_t* __restrict__ out,
                                     int Kdim, int Ndim) {
  int i = blockIdx.x * blockDim.x + threadIdx.x;
  int stride = gridDim.x * blockDim.x;
  int total = Kdim * Ndim;
  for (; i < total; i += stride) {
    int k = i / Ndim, n = i % Ndim;                 // coalesced read
    out[(size_t)n * Kdim + k] = f2bf(in[i]);
  }
}

// ---------------------------------------------------------------------------
// bf16 GEMM: out = A(MxK) * Wt(NxK)^T + bias.  Block tile 128x128, 8 waves
// (2x4), wave tile 64x32 = 4x2 WMMA tiles, K stepped 32 through LDS.
// Both 128x32 tiles are staged by the Tensor Data Mover with double
// buffering: wave 0 issues the next iteration's two DMAs, waits
// TENSORcnt<=2 (in-order completion => current buffer landed), barrier,
// all waves compute while the DMA engine fills the other buffer.
// store_mode: 0 = bf16 row-major, 1 = f32 row-major (final output),
//             2 = bf16 scattered to (B,H,D,T) for the attention V operand.
// ---------------------------------------------------------------------------
__launch_bounds__(256)
__global__ void gemm_bf16_kernel(const bf16_t* __restrict__ A,
                                 const bf16_t* __restrict__ Wt,
                                 const float* __restrict__ bias,
                                 void* __restrict__ out, int store_mode,
                                 int Kdim, int Ndim) {
  __shared__ __align__(32) bf16_t As[2][128 * 32];
  __shared__ __align__(32) bf16_t Bs[2][128 * 32];   // [n][k], W pre-transposed

  const int tid  = threadIdx.x;
  const int wid  = tid >> 5;
  const int lane = tid & 31;
  const int l15  = lane & 15;
  const int hi   = lane >> 4;               // which 16-lane half

  const int blockN = blockIdx.x * 128;
  const int blockM = blockIdx.y * 128;
  const int wm = (wid >> 2) * 64;           // wave M offset in block tile
  const int wn = (wid & 3) * 32;            // wave N offset in block tile

  unsigned int as_lds[2], bs_lds[2];
  as_lds[0] = (unsigned int)(uintptr_t)(void*)&As[0][0];
  as_lds[1] = (unsigned int)(uintptr_t)(void*)&As[1][0];
  bs_lds[0] = (unsigned int)(uintptr_t)(void*)&Bs[0][0];
  bs_lds[1] = (unsigned int)(uintptr_t)(void*)&Bs[1][0];

  const bf16_t* a_base = &A[(size_t)blockM * Kdim];
  const bf16_t* w_base = &Wt[(size_t)blockN * Kdim];

  v8f zero = {};
  v8f acc[4][2];
  for (int m = 0; m < 4; ++m)
    for (int n = 0; n < 2; ++n) acc[m][n] = zero;

  // preload tiles for kt = 0 into buffer 0
  if (wid == 0) {
    tdm_load_tile_bf16(a_base, as_lds[0], 128, 32, Kdim);
    tdm_load_tile_bf16(w_base, bs_lds[0], 128, 32, Kdim);
  }

  int cur = 0;
  for (int kt = 0; kt < Kdim; kt += 32, cur ^= 1) {
    __syncthreads();   // everyone done reading buffer cur^1 (previous iter)
    if (wid == 0) {
      if (kt + 32 < Kdim) {
        tdm_load_tile_bf16(a_base + kt + 32, as_lds[cur ^ 1], 128, 32, Kdim);
        tdm_load_tile_bf16(w_base + kt + 32, bs_lds[cur ^ 1], 128, 32, Kdim);
        __builtin_amdgcn_s_wait_tensorcnt(2);   // older two (buffer cur) landed
      } else {
        __builtin_amdgcn_s_wait_tensorcnt(0);
      }
    }
    __syncthreads();   // buffer cur is ready for all waves

    v16bf af[4], bfr[2];
#pragma unroll
    for (int m = 0; m < 4; ++m) {
      int row = wm + m * 16 + l15;
      af[m] = *reinterpret_cast<const v16bf*>(&As[cur][row * 32 + hi * 16]);
    }
#pragma unroll
    for (int n = 0; n < 2; ++n) {
      int col = wn + n * 16 + l15;
      bfr[n] = *reinterpret_cast<const v16bf*>(&Bs[cur][col * 32 + hi * 16]);
    }
#pragma unroll
    for (int m = 0; m < 4; ++m)
#pragma unroll
      for (int n = 0; n < 2; ++n)
        acc[m][n] = __builtin_amdgcn_wmma_f32_16x16x32_bf16(
            false, af[m], false, bfr[n], (short)0, acc[m][n], false, false);
  }

  // epilogue: C/D layout -> M = v + 8*hi, N = l15
#pragma unroll
  for (int m = 0; m < 4; ++m) {
#pragma unroll
    for (int n = 0; n < 2; ++n) {
      int gcol = blockN + wn + n * 16 + l15;
      float bv = bias[gcol];
#pragma unroll
      for (int v = 0; v < 8; ++v) {
        int grow = blockM + wm + m * 16 + v + hi * 8;
        float val = acc[m][n][v] + bv;
        if (store_mode == 1) {
          reinterpret_cast<float*>(out)[(size_t)grow * Ndim + gcol] = val;
        } else if (store_mode == 0) {
          reinterpret_cast<bf16_t*>(out)[(size_t)grow * Ndim + gcol] = f2bf(val);
        } else {  // V operand: (B,T,(H,D)) -> (B,H,D,T)
          int t = grow & (TT - 1), bb = grow >> 11;
          int hh = gcol >> 6,     dd = gcol & 63;
          reinterpret_cast<bf16_t*>(out)[(((size_t)(bb * HH + hh)) * DD + dd) * TT + t] = f2bf(val);
        }
      }
    }
  }
}

// ---------------------------------------------------------------------------
// Flash-style causal attention.  Q/K bf16 in (B,T,C), V bf16 in (B,H,D,T).
// Grid (T/128, H, B); 8 waves/block, one 16-row query tile per wave
// (no block sync -> per-wave causal loop lengths are safe).
// Per 32-key tile: S = Q*K^T (4 WMMAs), online softmax with width-16
// shfl_xor reductions, P staged bf16 through per-wave LDS, O += P*V
// (4 WMMAs, V fragments are single contiguous 32B loads thanks to layout).
// ---------------------------------------------------------------------------
__launch_bounds__(256)
__global__ void attn_kernel(const bf16_t* __restrict__ Q,
                            const bf16_t* __restrict__ K,
                            const bf16_t* __restrict__ Vt,
                            bf16_t* __restrict__ O) {
  __shared__ __align__(32) bf16_t Plds[8 * 16 * 32];  // per-wave 16x32 P tile

  const int tid  = threadIdx.x;
  const int wid  = tid >> 5;
  const int lane = tid & 31;
  const int l15  = lane & 15;
  const int hi   = lane >> 4;
  const int h = blockIdx.y, b = blockIdx.z;
  const int qbase = (blockIdx.x * 8 + wid) * 16;
  const size_t base_bt = (size_t)(b * TT) * CC + (size_t)h * DD;   // Q/K rows
  const size_t vhead   = (size_t)(b * HH + h) * DD;                // Vt head base
  bf16_t* pw = &Plds[wid * 512];

  // Q fragments: 16 rows x D=64 -> two K-slices of 32
  v16bf qf[2];
  {
    size_t qrow = base_bt + (size_t)(qbase + l15) * CC;
    qf[0] = *reinterpret_cast<const v16bf*>(&Q[qrow + hi * 16]);
    qf[1] = *reinterpret_cast<const v16bf*>(&Q[qrow + 32 + hi * 16]);
  }

  v8f zero = {};
  v8f o[4];
  for (int d = 0; d < 4; ++d) o[d] = zero;
  float mrun[8], lrun[8];
#pragma unroll
  for (int v = 0; v < 8; ++v) { mrun[v] = -1e30f; lrun[v] = 0.f; }

  const int ntiles = (qbase + 47) / 32;  // 32-key tiles covering keys <= qbase+15
  for (int j = 0; j < ntiles; ++j) {
    const int kb = j * 32;

    // ---- S = Q * K^T : 16 rows x 32 keys (two 16x16 N-tiles, K=64 chained)
    v8f s[2];
#pragma unroll
    for (int n = 0; n < 2; ++n) {
      size_t krow = base_bt + (size_t)(kb + n * 16 + l15) * CC;
      v16bf kf0 = *reinterpret_cast<const v16bf*>(&K[krow + hi * 16]);
      v16bf kf1 = *reinterpret_cast<const v16bf*>(&K[krow + 32 + hi * 16]);
      v8f c = zero;
      c = __builtin_amdgcn_wmma_f32_16x16x32_bf16(false, qf[0], false, kf0, (short)0, c, false, false);
      c = __builtin_amdgcn_wmma_f32_16x16x32_bf16(false, qf[1], false, kf1, (short)0, c, false, false);
      s[n] = c;
    }

    // ---- scale (1/sqrt(64)) + causal mask in the C layout
#pragma unroll
    for (int n = 0; n < 2; ++n) {
      int key = kb + n * 16 + l15;
#pragma unroll
      for (int v = 0; v < 8; ++v) {
        int row = qbase + v + hi * 8;
        float x = s[n][v] * 0.125f;
        s[n][v] = (key > row) ? -1e30f : x;
      }
    }

    // ---- online softmax; all 16 lanes of a half share the same M row
#pragma unroll
    for (int v = 0; v < 8; ++v) {
      float t = fmaxf(s[0][v], s[1][v]);
      for (int msk = 8; msk >= 1; msk >>= 1) t = fmaxf(t, __shfl_xor(t, msk, 16));
      float mn = fmaxf(mrun[v], t);
      float p0 = __expf(s[0][v] - mn);
      float p1 = __expf(s[1][v] - mn);
      float rs = p0 + p1;
      for (int msk = 8; msk >= 1; msk >>= 1) rs += __shfl_xor(rs, msk, 16);
      float alpha = __expf(mrun[v] - mn);
      lrun[v] = lrun[v] * alpha + rs;
      mrun[v] = mn;
#pragma unroll
      for (int d = 0; d < 4; ++d) o[d][v] = o[d][v] * alpha;
      int row = v + hi * 8;
      pw[row * 32 + l15]      = f2bf(p0);
      pw[row * 32 + 16 + l15] = f2bf(p1);
    }
    // cross-lane LDS dependency inside one wave: drain DScnt, pin ordering
    asm volatile("s_wait_dscnt 0" ::: "memory");
    __builtin_amdgcn_wave_barrier();

    // ---- O += P * V  (P: 16x32 bf16 A-operand from LDS; V: contiguous in T)
    v16bf pf = *reinterpret_cast<const v16bf*>(&pw[l15 * 32 + hi * 16]);
#pragma unroll
    for (int d = 0; d < 4; ++d) {
      int dcol = d * 16 + l15;
      v16bf vf = *reinterpret_cast<const v16bf*>(&Vt[(vhead + dcol) * TT + kb + hi * 16]);
      o[d] = __builtin_amdgcn_wmma_f32_16x16x32_bf16(false, pf, false, vf, (short)0, o[d], false, false);
    }
    // pf's ds_loads must land before the next iteration overwrites pw
    asm volatile("s_wait_dscnt 0" ::: "memory");
    __builtin_amdgcn_wave_barrier();
  }

  // ---- normalize and store (bf16, (B,T,C) layout for the out-projection)
#pragma unroll
  for (int d = 0; d < 4; ++d) {
    int dcol = d * 16 + l15;
#pragma unroll
    for (int v = 0; v < 8; ++v) {
      int row = qbase + v + hi * 8;
      float val = o[d][v] / lrun[v];
      O[base_bt + (size_t)row * CC + dcol] = f2bf(val);
    }
  }
}

// ---------------------------------------------------------------------------
extern "C" void kernel_launch(void* const* d_in, const int* in_sizes, int n_in,
                              void* d_out, int out_size, void* d_ws, size_t ws_size,
                              hipStream_t stream) {
  (void)in_sizes; (void)n_in; (void)out_size; (void)ws_size;
  const float* x  = (const float*)d_in[0];
  const float* Wq = (const float*)d_in[1];
  const float* bq = (const float*)d_in[2];
  const float* Wk = (const float*)d_in[3];
  const float* bk = (const float*)d_in[4];
  const float* Wv = (const float*)d_in[5];
  const float* bv = (const float*)d_in[6];
  const float* Wp = (const float*)d_in[7];
  const float* bp = (const float*)d_in[8];
  float* out = (float*)d_out;

  // workspace layout (bf16): x | WqT WkT WvT WpT | q k vT | attn_out (~88 MB)
  bf16_t* ws = (bf16_t*)d_ws;
  const size_t nx = (size_t)MM * CC;  // 8M elems
  const size_t nw = (size_t)CC * CC;  // 1M elems
  bf16_t* xb  = ws;
  bf16_t* wqb = xb  + nx;   // transposed (NxK)
  bf16_t* wkb = wqb + nw;
  bf16_t* wvb = wkb + nw;
  bf16_t* wpb = wvb + nw;
  bf16_t* qb  = wpb + nw;
  bf16_t* kb  = qb  + nx;
  bf16_t* vtb = kb  + nx;   // (B,H,D,T)
  bf16_t* ab  = vtb + nx;

  f32_to_bf16_kernel<<<2048, 256, 0, stream>>>(x, xb, (int)nx);
  f32_to_bf16_T_kernel<<<1024, 256, 0, stream>>>(Wq, wqb, CC, CC);
  f32_to_bf16_T_kernel<<<1024, 256, 0, stream>>>(Wk, wkb, CC, CC);
  f32_to_bf16_T_kernel<<<1024, 256, 0, stream>>>(Wv, wvb, CC, CC);
  f32_to_bf16_T_kernel<<<1024, 256, 0, stream>>>(Wp, wpb, CC, CC);

  dim3 gg(CC / 128, MM / 128);  // (8, 64)
  gemm_bf16_kernel<<<gg, 256, 0, stream>>>(xb, wqb, bq, qb,  0, CC, CC);
  gemm_bf16_kernel<<<gg, 256, 0, stream>>>(xb, wkb, bk, kb,  0, CC, CC);
  gemm_bf16_kernel<<<gg, 256, 0, stream>>>(xb, wvb, bv, vtb, 2, CC, CC);

  dim3 ga(TT / 128, HH, BB);    // (16, 16, 4)
  attn_kernel<<<ga, 256, 0, stream>>>(qb, kb, vtb, ab);

  gemm_bf16_kernel<<<gg, 256, 0, stream>>>(ab, wpb, bp, out, 1, CC, CC);
}
